// PointNet2_FBS_SSG_23922967838989
// MI455X (gfx1250) — matrix operand order repaired and checked by reading, not compiled, not run on hardware
//
#include <hip/hip_runtime.h>

typedef __attribute__((ext_vector_type(16))) _Float16 v16h;
typedef __attribute__((ext_vector_type(8)))  _Float16 v8h;
typedef __attribute__((ext_vector_type(8)))  float    v8f;
typedef __attribute__((ext_vector_type(4)))  int      v4i;

#define BN_EPS_F 1e-5f

constexpr int   B_     = 4;
constexpr int   N_     = 16384;
constexpr int   CIN_   = 128;
constexpr int   S_     = 1024;   // NPOINT
constexpr int   K_     = 32;     // NSAMPLE
constexpr int   FG_    = 512;    // FG_NSAMPLE
constexpr int   TOPKN_ = 2048;
constexpr float R2_    = 0.3f * 0.3f;
constexpr int   NBINS_ = 8192;

// -------------------- async global -> LDS staging (gfx1250) -------------------
#if __has_builtin(__builtin_amdgcn_global_load_async_to_lds_b128)
#define HAVE_ASYNC_LDS 1
#else
#define HAVE_ASYNC_LDS 0
#endif

#if HAVE_ASYNC_LDS
// exact builtin param types: (v4i addrspace(1)*, v4i addrspace(3)*, imm int, imm int)
typedef v4i __attribute__((address_space(1)))* gp1_t;
typedef v4i __attribute__((address_space(3)))* lp3_t;

static __device__ __forceinline__ void wait_async_all() {
#if __has_builtin(__builtin_amdgcn_s_wait_asynccnt)
  __builtin_amdgcn_s_wait_asynccnt(0);
#else
  asm volatile("s_wait_asynccnt 0x0" ::: "memory");
#endif
}
#endif

// Stage N_*3 floats (one batch of xyz, 192KB) into LDS cooperatively.
static __device__ __forceinline__ void stage_xyz_to_lds(const float* __restrict__ x,
                                                        float* lx, int t, int nthreads)
{
#if HAVE_ASYNC_LDS
  unsigned long long g = (unsigned long long)(const void*)x;
  unsigned int       l = (unsigned int)(unsigned long long)(void*)lx;  // flat->LDS offset (low 32b)
  const int chunks = (N_ * 3) / 4;     // 16B chunks
  for (int c = t; c < chunks; c += nthreads) {
    __builtin_amdgcn_global_load_async_to_lds_b128(
        (gp1_t)(g + (unsigned long long)c * 16u),
        (lp3_t)(unsigned long long)(l + (unsigned int)c * 16u),
        0, 0);
  }
  wait_async_all();
#else
  for (int f = t; f < N_ * 3; f += nthreads) lx[f] = x[f];
#endif
  __syncthreads();
}

// =====================================================================
// WMMA GEMM:  Y[r, o] = sum_c A[o, c] * X[r, c]
//   A: [O, Cpad] f16 row-major (weights)       X: [Rows, Cpad] f16
//   Y: [Rows, O]  f32 row-major
// One wave computes one 16(o) x 64(rows) block = 4 accumulators: the A
// fragment is loaded once per k-step and feeds 4 INDEPENDENT WMMAs
// (deep XDL pipelining, 2.5 VMEM per WMMA instead of 4).
// A-fragment (16-bit A 16x32): lane L elem i -> K=(L/16)*16+i
// B-fragment (16-bit B 32x16): lane L elem i -> K=(L/16)*16+i
// D: VGPR r, lanes 0-15 -> M=r, lanes 16-31 -> M=8+r  (contiguous v8f store)
// =====================================================================
__global__ __launch_bounds__(256) void gemm_wmma_f16(
    const _Float16* __restrict__ A,
    const _Float16* __restrict__ X,
    float* __restrict__ Y,
    int O, int Cpad, int Rows)
{
  const int lane = threadIdx.x & 31;
  const int wave = blockIdx.x * (blockDim.x >> 5) + (threadIdx.x >> 5);
  const int tiles_o  = O >> 4;
  const int groups_n = Rows >> 6;                 // 4 row-tiles per wave
  if (wave >= tiles_o * groups_n) return;         // wave-uniform
  const int to = wave % tiles_o;
  const int tg = wave / tiles_o;

  const int half_sel = lane >> 4;                 // 0 or 1
  const int lane16   = lane & 15;

  const _Float16* ap = A + (size_t)((to << 4) + lane16) * Cpad + half_sel * 16;
  const size_t row0 = ((size_t)tg << 6) + lane16;
  const _Float16* bp0 = X + (row0 +  0) * Cpad + half_sel * 16;
  const _Float16* bp1 = X + (row0 + 16) * Cpad + half_sel * 16;
  const _Float16* bp2 = X + (row0 + 32) * Cpad + half_sel * 16;
  const _Float16* bp3 = X + (row0 + 48) * Cpad + half_sel * 16;

  v8f acc0 = {}, acc1 = {}, acc2 = {}, acc3 = {};
  for (int k0 = 0; k0 < Cpad; k0 += 32) {
    v8h alo = *(const v8h*)(ap + k0);
    v8h ahi = *(const v8h*)(ap + k0 + 8);
    v16h a = __builtin_shufflevector(alo, ahi,
        0,1,2,3,4,5,6,7,8,9,10,11,12,13,14,15);
    v16h b0 = *(const v16h*)(bp0 + k0);
    v16h b1 = *(const v16h*)(bp1 + k0);
    v16h b2 = *(const v16h*)(bp2 + k0);
    v16h b3 = *(const v16h*)(bp3 + k0);
    __builtin_prefetch(bp0 + k0 + 128, 0, 1);
    __builtin_prefetch(bp2 + k0 + 128, 0, 1);
    acc0 = __builtin_amdgcn_wmma_f32_16x16x32_f16(false, a, false, b0, (short)0, acc0, false, false);
    acc1 = __builtin_amdgcn_wmma_f32_16x16x32_f16(false, a, false, b1, (short)0, acc1, false, false);
    acc2 = __builtin_amdgcn_wmma_f32_16x16x32_f16(false, a, false, b2, (short)0, acc2, false, false);
    acc3 = __builtin_amdgcn_wmma_f32_16x16x32_f16(false, a, false, b3, (short)0, acc3, false, false);
  }
  const int ocol = (to << 4) + half_sel * 8;
  *(v8f*)(Y + (row0 +  0) * O + ocol) = acc0;
  *(v8f*)(Y + (row0 + 16) * O + ocol) = acc1;
  *(v8f*)(Y + (row0 + 32) * O + ocol) = acc2;
  *(v8f*)(Y + (row0 + 48) * O + ocol) = acc3;
}

// ------------------- weight repack f32 -> f16 with zero pad -------------------
__global__ void pack_weight_f16(const float* __restrict__ W, _Float16* __restrict__ Wh,
                                int O, int C, int Cpad)
{
  int i = blockIdx.x * blockDim.x + threadIdx.x;
  if (i >= O * Cpad) return;
  int c = i % Cpad, o = i / Cpad;
  Wh[i] = (c < C) ? (_Float16)W[o * C + c] : (_Float16)0.0f;
}

// ------------- features [B,C,N] f32 -> F [(b*N+n), C] f16 point-major ---------
__global__ void pack_features_f16(const float* __restrict__ feat, _Float16* __restrict__ F)
{
  size_t i = (size_t)blockIdx.x * blockDim.x + threadIdx.x;
  size_t total = (size_t)B_ * N_ * CIN_;
  if (i >= total) return;
  int c = (int)(i % CIN_);
  int n = (int)((i / CIN_) % N_);
  int b = (int)(i / ((size_t)CIN_ * N_));
  F[i] = (_Float16)feat[((size_t)b * CIN_ + c) * N_ + n];
}

// --------------------- per-channel BN statistics over rows --------------------
__global__ __launch_bounds__(256) void bn_stats(const float* __restrict__ Y,
                                                float* __restrict__ mean,
                                                float* __restrict__ var,
                                                int O, int Rows)
{
  __shared__ float ss[256], sq[256];
  const int ch = blockIdx.x;
  float s = 0.f, q = 0.f;
  for (int r = threadIdx.x; r < Rows; r += 256) {
    float v = Y[(size_t)r * O + ch];
    s += v; q += v * v;
  }
  ss[threadIdx.x] = s; sq[threadIdx.x] = q;
  __syncthreads();
  for (int st = 128; st > 0; st >>= 1) {
    if (threadIdx.x < st) {
      ss[threadIdx.x] += ss[threadIdx.x + st];
      sq[threadIdx.x] += sq[threadIdx.x + st];
    }
    __syncthreads();
  }
  if (threadIdx.x == 0) {
    float mu = ss[0] / (float)Rows;
    mean[ch] = mu;
    var[ch]  = sq[0] / (float)Rows - mu * mu;
  }
}

// ----------------- normalize + ReLU, repack to f16 [Rows, Cpad] ---------------
__global__ void bn_relu_f16(const float* __restrict__ Y, _Float16* __restrict__ Xn,
                            const float* __restrict__ mean, const float* __restrict__ var,
                            const float* __restrict__ g, const float* __restrict__ bb,
                            int O, int Cpad, int Rows)
{
  size_t i = (size_t)blockIdx.x * blockDim.x + threadIdx.x;
  if (i >= (size_t)Rows * Cpad) return;
  int c = (int)(i % Cpad);
  size_t r = i / Cpad;
  float v = 0.f;
  if (c < O) {
    v = (Y[r * O + c] - mean[c]) * rsqrtf(var[c] + BN_EPS_F) * g[c] + bb[c];
    v = fmaxf(v, 0.f);
  }
  Xn[i] = (_Float16)v;
}

// ----------- mask scores (2x64 dot per point), softmax margin -----------------
__global__ void mask_margin_kernel(const _Float16* __restrict__ H,
                                   const float* __restrict__ fw2,
                                   const float* __restrict__ fbias2,
                                   float* __restrict__ out_scores, // [B,2,N]
                                   float* __restrict__ margin)     // [B,N]
{
  int i = blockIdx.x * blockDim.x + threadIdx.x;
  if (i >= B_ * N_) return;
  int n = i % N_, b = i / N_;
  const _Float16* h = H + (size_t)i * 64;
  float s0 = fbias2[0], s1 = fbias2[1];
  for (int c = 0; c < 64; ++c) {
    float hv = (float)h[c];
    s0 += fw2[c] * hv;
    s1 += fw2[64 + c] * hv;
  }
  out_scores[((size_t)b * 2 + 0) * N_ + n] = s0;
  out_scores[((size_t)b * 2 + 1) * N_ + n] = s1;
  float m = fmaxf(s0, s1);
  float e0 = __expf(s0 - m), e1 = __expf(s1 - m);
  margin[i] = (e1 - e0) / (e0 + e1);
}

// ------------------ histogram-based top-k threshold select --------------------
__global__ void hist_build(const float* __restrict__ margin, int* __restrict__ hist)
{
  int i = blockIdx.x * blockDim.x + threadIdx.x;
  if (i >= B_ * N_) return;
  int b = i / N_;
  float m = margin[i];                       // in (-1, 1)
  int bin = (int)((m + 1.0f) * 0.5f * (float)NBINS_);
  bin = min(max(bin, 0), NBINS_ - 1);
  atomicAdd(&hist[b * NBINS_ + bin], 1);
}

__global__ void hist_thresh(const int* __restrict__ hist, float* __restrict__ thresh)
{
  int b = blockIdx.x;
  if (threadIdx.x != 0) return;
  int acc = 0, sel = 0;
  for (int i = NBINS_ - 1; i >= 0; --i) {
    acc += hist[b * NBINS_ + i];
    if (acc >= TOPKN_) { sel = i; break; }
  }
  thresh[b] = -1.0f + 2.0f * (float)sel / (float)NBINS_;
}

__global__ void mask_build(const float* __restrict__ margin,
                           const float* __restrict__ thresh,
                           unsigned char* __restrict__ fg)
{
  int i = blockIdx.x * blockDim.x + threadIdx.x;
  if (i >= B_ * N_) return;
  int b = i / N_;
  fg[i] = margin[i] >= thresh[b] ? 1 : 0;
}

// ------------------------- masked farthest point sampling ---------------------
// One workgroup per batch. The batch's xyz (192KB) is async-staged into LDS and
// the running min-distance array (64KB) also lives in LDS -> 320KB WGP LDS pays
// off: all 512 sequential rounds run on LDS only.
__global__ __launch_bounds__(256) void fps_kernel(const float* __restrict__ xyz,
                                                  const unsigned char* __restrict__ fg,
                                                  int invert,
                                                  int* __restrict__ out_idx,  // [B,1024]
                                                  int nsel, int offset)
{
  extern __shared__ char smem[];
  float* lx = (float*)smem;                          // [N_*3]
  float* md = (float*)(smem + (size_t)N_ * 3 * 4);   // [N_]
  const int b = blockIdx.x;
  const int t = threadIdx.x;
  __shared__ float sval[256];
  __shared__ int   sidx[256];
  __shared__ float spt[3];

  stage_xyz_to_lds(xyz + (size_t)b * N_ * 3, lx, t, 256);

  for (int j = t; j < N_; j += 256) {
    bool ok = (fg[b * N_ + j] != 0) != (invert != 0);
    md[j] = ok ? 1e10f : -1.0f;
  }
  __syncthreads();

  for (int it = 0; it < nsel; ++it) {
    float bv = -2.0f; int bi = 0x7fffffff;
    for (int j = t; j < N_; j += 256) {
      float v = md[j];
      if (v > bv || (v == bv && j < bi)) { bv = v; bi = j; }
    }
    sval[t] = bv; sidx[t] = bi;
    __syncthreads();
    for (int st = 128; st > 0; st >>= 1) {
      if (t < st) {
        if (sval[t + st] > sval[t] ||
            (sval[t + st] == sval[t] && sidx[t + st] < sidx[t])) {
          sval[t] = sval[t + st]; sidx[t] = sidx[t + st];
        }
      }
      __syncthreads();
    }
    int sel = sidx[0];
    if (t == 0) {
      out_idx[b * S_ + offset + it] = sel;
      spt[0] = lx[sel * 3 + 0]; spt[1] = lx[sel * 3 + 1]; spt[2] = lx[sel * 3 + 2];
    }
    __syncthreads();
    float px = spt[0], py = spt[1], pz = spt[2];
    for (int j = t; j < N_; j += 256) {
      float m = md[j];
      if (m > -0.5f) {
        float dx = lx[j * 3 + 0] - px;
        float dy = lx[j * 3 + 1] - py;
        float dz = lx[j * 3 + 2] - pz;
        md[j] = fminf(m, dx * dx + dy * dy + dz * dz);
      }
    }
    __syncthreads();
  }
}

// ----------------- gather new_xyz + write indices (as float) ------------------
__global__ void gather_kernel(const float* __restrict__ xyz,
                              const int* __restrict__ indices,
                              float* __restrict__ out_newxyz,   // [B,S,3]
                              float* __restrict__ out_idx_f)    // [B,S]
{
  int i = blockIdx.x * blockDim.x + threadIdx.x;
  if (i >= B_ * S_) return;
  int b = i / S_;
  int id = indices[i];
  out_newxyz[(size_t)i * 3 + 0] = xyz[((size_t)b * N_ + id) * 3 + 0];
  out_newxyz[(size_t)i * 3 + 1] = xyz[((size_t)b * N_ + id) * 3 + 1];
  out_newxyz[(size_t)i * 3 + 2] = xyz[((size_t)b * N_ + id) * 3 + 2];
  out_idx_f[i] = (float)id;
}

// --------------------------------- ball query ---------------------------------
// Each 256-thread block handles 256 queries of one batch; the batch xyz is
// async-staged into LDS once and each thread scans from LDS with early exit.
__global__ __launch_bounds__(256) void ball_query_kernel(const float* __restrict__ xyz,
                                                         const float* __restrict__ new_xyz,
                                                         int* __restrict__ idx) // [B,S,K]
{
  extern __shared__ char smem[];
  float* lx = (float*)smem;                          // [N_*3]
  const int i = blockIdx.x * 256 + threadIdx.x;      // query id, block-uniform batch
  const int b = i / S_;

  stage_xyz_to_lds(xyz + (size_t)b * N_ * 3, lx, threadIdx.x, 256);

  float qx = new_xyz[(size_t)i * 3 + 0];
  float qy = new_xyz[(size_t)i * 3 + 1];
  float qz = new_xyz[(size_t)i * 3 + 2];
  int* o = idx + (size_t)i * K_;
  int cnt = 0;
  for (int j = 0; j < N_ && cnt < K_; ++j) {
    float dx = lx[j * 3 + 0] - qx;
    float dy = lx[j * 3 + 1] - qy;
    float dz = lx[j * 3 + 2] - qz;
    if (dx * dx + dy * dy + dz * dz <= R2_) o[cnt++] = j;
  }
  if (cnt == 0) {
    for (int k = 0; k < K_; ++k) o[k] = N_ - 1;
  } else {
    int first = o[0];
    for (int k = cnt; k < K_; ++k) o[k] = first;     // pad with first neighbor
  }
}

// ------------- build grouped input G[(b,s,k), 160] f16 point-major ------------
__global__ void group_kernel(const float* __restrict__ xyz,
                             const float* __restrict__ feats,
                             const float* __restrict__ new_xyz,
                             const int* __restrict__ idx,
                             _Float16* __restrict__ G, int Cpad)
{
  int i = blockIdx.x * blockDim.x + threadIdx.x;
  if (i >= B_ * S_ * K_) return;
  int s = (i / K_) % S_;
  int b = i / (K_ * S_);
  int j = idx[i];
  _Float16* g = G + (size_t)i * Cpad;
  g[0] = (_Float16)(xyz[((size_t)b * N_ + j) * 3 + 0] - new_xyz[((size_t)b * S_ + s) * 3 + 0]);
  g[1] = (_Float16)(xyz[((size_t)b * N_ + j) * 3 + 1] - new_xyz[((size_t)b * S_ + s) * 3 + 1]);
  g[2] = (_Float16)(xyz[((size_t)b * N_ + j) * 3 + 2] - new_xyz[((size_t)b * S_ + s) * 3 + 2]);
  const float* f = feats + (size_t)b * CIN_ * N_ + j;
  for (int c = 0; c < CIN_; ++c) g[3 + c] = (_Float16)f[(size_t)c * N_];
  for (int c = 3 + CIN_; c < Cpad; ++c) g[c] = (_Float16)0.0f;
}

// --------------- fused BN + ReLU + maxpool over nsample (layer 3) -------------
__global__ void bn_relu_maxpool(const float* __restrict__ Y,  // [(b,s,k), 256]
                                const float* __restrict__ mean, const float* __restrict__ var,
                                const float* __restrict__ g, const float* __restrict__ bb,
                                float* __restrict__ out_nf)   // [B, 256, S]
{
  const int O = 256;
  int i = blockIdx.x * blockDim.x + threadIdx.x;
  if (i >= B_ * S_ * O) return;
  int o = i % O;
  int s = (i / O) % S_;
  int b = i / (O * S_);
  float mu = mean[o];
  float is = rsqrtf(var[o] + BN_EPS_F) * g[o];
  float be = bb[o];
  float best = -1e30f;
  size_t base = ((size_t)b * S_ + s) * K_;
  for (int k = 0; k < K_; ++k) {
    float v = (Y[(base + k) * O + o] - mu) * is + be;
    v = fmaxf(v, 0.f);
    best = fmaxf(best, v);
  }
  out_nf[((size_t)b * O + o) * S_ + s] = best;
}

// ==============================================================================
static inline int cdiv(long long a, long long b) { return (int)((a + b - 1) / b); }

extern "C" void kernel_launch(void* const* d_in, const int* in_sizes, int n_in,
                              void* d_out, int out_size, void* d_ws, size_t ws_size,
                              hipStream_t stream)
{
  (void)in_sizes; (void)n_in; (void)out_size; (void)ws_size;
  const float* xyz    = (const float*)d_in[0];
  const float* feats  = (const float*)d_in[1];
  const float* w1     = (const float*)d_in[2];
  const float* g1     = (const float*)d_in[3];
  const float* b1     = (const float*)d_in[4];
  const float* w2     = (const float*)d_in[5];
  const float* g2     = (const float*)d_in[6];
  const float* b2     = (const float*)d_in[7];
  const float* w3     = (const float*)d_in[8];
  const float* g3     = (const float*)d_in[9];
  const float* b3     = (const float*)d_in[10];
  const float* fw1    = (const float*)d_in[11];
  const float* fg1    = (const float*)d_in[12];
  const float* fb1    = (const float*)d_in[13];
  const float* fw2    = (const float*)d_in[14];
  const float* fbias2 = (const float*)d_in[15];
  float* out = (float*)d_out;

  // output regions (flat concat of tuple)
  float* out_newxyz = out;                              // 4*1024*3
  float* out_nf     = out + 12288;                      // 4*256*1024
  float* out_idx_f  = out + 12288 + 1048576;            // 4*1024
  float* out_mask   = out + 12288 + 1048576 + 4096;     // 4*2*16384

  // workspace bump allocator (256B aligned)
  char* ws = (char*)d_ws;
  size_t off = 0;
  auto alloc = [&](size_t bytes) {
    size_t r = off; off = (off + bytes + 255) & ~(size_t)255; return r;
  };
  const int ROWS1 = B_ * N_;          // 65536
  const int ROWS2 = B_ * S_ * K_;     // 131072
  const int CP1 = 160;                // conv1 Cpad (131 -> 160)

  size_t oY    = alloc((size_t)ROWS2 * 256 * 4);        // big f32 Y (reused all GEMMs)
  size_t oG    = alloc((size_t)ROWS2 * CP1 * 2);        // G f16; reused as X2
  size_t oX3   = alloc((size_t)ROWS2 * 128 * 2);        // X3 f16; early: F + H
  size_t oMarg = alloc((size_t)ROWS1 * 4);
  size_t oHist = alloc((size_t)B_ * NBINS_ * 4);
  size_t oThr  = alloc(256);
  size_t oFg   = alloc((size_t)ROWS1);
  size_t oSel  = alloc((size_t)B_ * S_ * 4);
  size_t oBall = alloc((size_t)B_ * S_ * K_ * 4);
  size_t oMean = alloc(256 * 4);
  size_t oVar  = alloc(256 * 4);
  size_t oWf1  = alloc((size_t)64 * 128 * 2);
  size_t oW1   = alloc((size_t)128 * CP1 * 2);
  size_t oW2   = alloc((size_t)128 * 128 * 2);
  size_t oW3   = alloc((size_t)256 * 128 * 2);

  float*         Ybuf  = (float*)(ws + oY);
  float*         Hy    = (float*)(ws + oY);                       // reuse of Y region
  _Float16*      G     = (_Float16*)(ws + oG);
  _Float16*      X2    = (_Float16*)(ws + oG);                    // reuse after GEMM2
  _Float16*      X3    = (_Float16*)(ws + oX3);
  _Float16*      F     = (_Float16*)(ws + oX3);                   // early reuse
  _Float16*      H     = (_Float16*)(ws + oX3 + (size_t)ROWS1 * 128 * 2);
  float*         marg  = (float*)(ws + oMarg);
  int*           hist  = (int*)(ws + oHist);
  float*         thr   = (float*)(ws + oThr);
  unsigned char* fg    = (unsigned char*)(ws + oFg);
  int*           sel   = (int*)(ws + oSel);
  int*           ball  = (int*)(ws + oBall);
  float*         mean  = (float*)(ws + oMean);
  float*         var   = (float*)(ws + oVar);
  _Float16*      Wf1h  = (_Float16*)(ws + oWf1);
  _Float16*      W1h   = (_Float16*)(ws + oW1);
  _Float16*      W2h   = (_Float16*)(ws + oW2);
  _Float16*      W3h   = (_Float16*)(ws + oW3);

  const int TB = 256;
  auto gemm = [&](const _Float16* A, const _Float16* X, float* Y,
                  int O, int Cpad, int Rows) {
    int waves = (O >> 4) * (Rows >> 6);     // 4 row-tiles per wave
    gemm_wmma_f16<<<cdiv(waves, 8), TB, 0, stream>>>(A, X, Y, O, Cpad, Rows);
  };

  // 0) repack weights to f16 (zero-padded K)
  pack_weight_f16<<<cdiv(64 * 128, TB), TB, 0, stream>>>(fw1, Wf1h, 64, 128, 128);
  pack_weight_f16<<<cdiv(128 * CP1, TB), TB, 0, stream>>>(w1, W1h, 128, 131, CP1);
  pack_weight_f16<<<cdiv(128 * 128, TB), TB, 0, stream>>>(w2, W2h, 128, 128, 128);
  pack_weight_f16<<<cdiv(256 * 128, TB), TB, 0, stream>>>(w3, W3h, 256, 128, 128);

  // 1) features -> point-major f16 ; h = relu(BN(fw1 @ features))
  pack_features_f16<<<cdiv((long long)ROWS1 * CIN_, TB), TB, 0, stream>>>(feats, F);
  gemm(Wf1h, F, Hy, 64, 128, ROWS1);
  bn_stats<<<64, 256, 0, stream>>>(Hy, mean, var, 64, ROWS1);
  bn_relu_f16<<<cdiv((long long)ROWS1 * 64, TB), TB, 0, stream>>>(
      Hy, H, mean, var, fg1, fb1, 64, 64, ROWS1);

  // 2) mask scores + softmax margin; histogram top-k threshold; fg mask
  mask_margin_kernel<<<cdiv(ROWS1, TB), TB, 0, stream>>>(H, fw2, fbias2, out_mask, marg);
  (void)hipMemsetAsync(hist, 0, (size_t)B_ * NBINS_ * 4, stream);
  hist_build<<<cdiv(ROWS1, TB), TB, 0, stream>>>(marg, hist);
  hist_thresh<<<B_, 32, 0, stream>>>(hist, thr);
  mask_build<<<cdiv(ROWS1, TB), TB, 0, stream>>>(marg, thr, fg);

  // 3) masked FPS: 512 fg + 512 bg (xyz + mind fully in LDS, async-staged)
  const size_t FPS_SMEM  = (size_t)N_ * 3 * 4 + (size_t)N_ * 4;   // 256KB
  const size_t BALL_SMEM = (size_t)N_ * 3 * 4;                    // 192KB
  fps_kernel<<<B_, 256, FPS_SMEM, stream>>>(xyz, fg, 0, sel, FG_, 0);
  fps_kernel<<<B_, 256, FPS_SMEM, stream>>>(xyz, fg, 1, sel, S_ - FG_, FG_);

  // 4) gather new_xyz (+ indices as float), ball query, group
  gather_kernel<<<cdiv(B_ * S_, TB), TB, 0, stream>>>(xyz, sel, out_newxyz, out_idx_f);
  ball_query_kernel<<<B_ * S_ / 256, 256, BALL_SMEM, stream>>>(xyz, out_newxyz, ball);
  group_kernel<<<cdiv(ROWS2, TB), TB, 0, stream>>>(xyz, feats, out_newxyz, ball, G, CP1);

  // 5) conv1: [128x131] ; BN+ReLU -> X2 (reuses G region)
  gemm(W1h, G, Ybuf, 128, CP1, ROWS2);
  bn_stats<<<128, 256, 0, stream>>>(Ybuf, mean, var, 128, ROWS2);
  bn_relu_f16<<<cdiv((long long)ROWS2 * 128, TB), TB, 0, stream>>>(
      Ybuf, X2, mean, var, g1, b1, 128, 128, ROWS2);

  // 6) conv2: [128x128] ; BN+ReLU -> X3 (reuses F/H region)
  gemm(W2h, X2, Ybuf, 128, 128, ROWS2);
  bn_stats<<<128, 256, 0, stream>>>(Ybuf, mean, var, 128, ROWS2);
  bn_relu_f16<<<cdiv((long long)ROWS2 * 128, TB), TB, 0, stream>>>(
      Ybuf, X3, mean, var, g2, b2, 128, 128, ROWS2);

  // 7) conv3: [256x128] ; fused BN+ReLU+maxpool -> new_features
  gemm(W3h, X3, Ybuf, 256, 128, ROWS2);
  bn_stats<<<256, 256, 0, stream>>>(Ybuf, mean, var, 256, ROWS2);
  bn_relu_maxpool<<<cdiv(B_ * S_ * 256, TB), TB, 0, stream>>>(
      Ybuf, mean, var, g3, b3, out_nf);
}
